// MPN_10900626998070
// MI455X (gfx1250) — matrix operand order repaired and checked by reading, not compiled, not run on hardware
//
#include <hip/hip_runtime.h>
#include <hip/hip_bf16.h>

// ---------------------------------------------------------------------------
// Types for CDNA5 WMMA (wave32): v_wmma_f32_16x16x32_bf16
// ---------------------------------------------------------------------------
typedef __attribute__((ext_vector_type(16))) __bf16       v16bf;
typedef __attribute__((ext_vector_type(8)))  float        v8f;
typedef __attribute__((ext_vector_type(8)))  unsigned int v8u;

#define HID 128

// fp32 -> bf16 via native convert (v_cvt_pk_bf16_f32 on gfx1250)
__device__ __forceinline__ __bf16 f2bf(float f) { return (__bf16)f; }

// Pack two fp32 into one dword of two bf16 (lo = first) - lowers to cvt_pk
__device__ __forceinline__ unsigned int pack_bf16x2(float lo, float hi) {
    unsigned short a = __builtin_bit_cast(unsigned short, f2bf(lo));
    unsigned short b = __builtin_bit_cast(unsigned short, f2bf(hi));
    return (unsigned int)a | ((unsigned int)b << 16);
}

struct U8x32 { uint4 lo, hi; };  // 32 bytes == one v16bf fragment

// ---------------------------------------------------------------------------
// WMMA fragment loaders (layouts per CDNA5 ISA 7.12.2, wave32).
//
// A (16x32 bf16) from LDS stored as packed (k,k+1) pairs, row-major:
// the 8 fragment dwords are two contiguous 4-dword chunks at pair offsets
// kb/2 + {0..3} and kb/2 + {8..11}  ->  two ds_load_b128.
// ---------------------------------------------------------------------------
__device__ __forceinline__ v16bf lds_a_frag128(const unsigned int* A32,
                                               int ldaPairs, int r0, int k0,
                                               int lane) {
    int m   = r0 + (lane & 15);
    int kb2 = (k0 >> 1) + ((lane >= 16) ? 4 : 0);
    const unsigned int* p = A32 + m * ldaPairs + kb2;
    U8x32 u;
    u.lo = *(const uint4*)(p);
    u.hi = *(const uint4*)(p + 8);
    return __builtin_bit_cast(v16bf, u);
}

// B (32x16 bf16) from LDS stored fragment-major: sw[tile][lane][8 dwords].
// One fragment = 8 contiguous dwords -> two ds_load_b128.
__device__ __forceinline__ v16bf lds_b_frag_sw(const unsigned int* sw,
                                               int tileId, int lane) {
    const unsigned int* p = sw + tileId * 256 + lane * 8;
    U8x32 u;
    u.lo = *(const uint4*)(p);
    u.hi = *(const uint4*)(p + 4);
    return __builtin_bit_cast(v16bf, u);
}

// A fragment from fp32 global activations (pairs contiguous -> b64 + cvt_pk)
__device__ __forceinline__ v16bf gl_a_frag_f32(const float* A, int lda, int r0,
                                               int k0, int lane) {
    v8u u;
    int m  = r0 + (lane & 15);
    int kb = k0 + ((lane >= 16) ? 8 : 0);
#pragma unroll
    for (int i = 0; i < 8; ++i) {
        int k = kb + ((i < 4) ? 2 * i : 2 * i + 8);
        const float* p = &A[(long)m * lda + k];
        u[i] = pack_bf16x2(p[0], p[1]);
    }
    return __builtin_bit_cast(v16bf, u);
}

__device__ __forceinline__ v8f wmma_bf16(v16bf a, v16bf b, v8f c) {
    return __builtin_amdgcn_wmma_f32_16x16x32_bf16(false, a, false, b,
                                                   (short)0, c, false, false);
}

// Swizzled-store helper: given source pair-row pk (= k/2) and column n of a
// [K,128] fp32 weight matrix, compute position in sw[tile][lane][j].
// tile enumerates (kstep, coltile) as tileId = (pk>>4)*8 + (n>>4).
__device__ __forceinline__ int b_sw_index(int pk, int n) {
    int tileId = (pk >> 4) * 8 + (n >> 4);
    int lane2  = (n & 15) + (((pk >> 2) & 1) << 4);
    int j      = (pk & 3) + (((pk >> 3) & 1) << 2);
    return tileId * 256 + lane2 * 8 + j;
}

// ---------------------------------------------------------------------------
// Small helper kernels
// ---------------------------------------------------------------------------
__global__ void zero_kernel(float* __restrict__ p, long n) {
    long i = (long)blockIdx.x * blockDim.x + threadIdx.x;
    long s = (long)gridDim.x * blockDim.x;
    for (; i < n; i += s) p[i] = 0.f;
}

__global__ void deg_kernel(const int* __restrict__ col, float* __restrict__ deg,
                           int E) {
    int e = blockIdx.x * blockDim.x + threadIdx.x;
    if (e < E) atomicAdd(&deg[col[e]], 1.0f);
}

__global__ void dinv_kernel(float* __restrict__ deg, int N) {
    int i = blockIdx.x * blockDim.x + threadIdx.x;
    if (i < N) {
        float d = deg[i];
        deg[i] = (d > 0.f) ? (1.0f / sqrtf(d)) : 0.f;
    }
}

__global__ void norm_kernel(const int* __restrict__ row,
                            const int* __restrict__ col,
                            const float* __restrict__ dinv,
                            float* __restrict__ norm, int E) {
    int e = blockIdx.x * blockDim.x + threadIdx.x;
    if (e < E) norm[e] = dinv[row[e]] * dinv[col[e]];
}

// SpMM: pout[col[e]] += norm[e] * pin[row[e]]   (one thread per edge x 4 cols)
__global__ void spmm_kernel(const int* __restrict__ row,
                            const int* __restrict__ col,
                            const float* __restrict__ norm,
                            const float* __restrict__ pin,
                            float* __restrict__ pout, int E) {
    long tid = (long)blockIdx.x * blockDim.x + threadIdx.x;
    long e = tid >> 5;
    if (e >= E) return;
    int g = (int)(tid & 31);
    int r = row[e], c = col[e];
    float nv = norm[e];
    const float4 v = *((const float4*)(pin + (long)r * HID) + g);
    float* po = pout + (long)c * HID + g * 4;
    atomicAdd(po + 0, nv * v.x);
    atomicAdd(po + 1, nv * v.y);
    atomicAdd(po + 2, nv * v.z);
    atomicAdd(po + 3, nv * v.w);
}

// ---------------------------------------------------------------------------
// Edge MLP: m = relu([x_i, x_j, e] @ W1 + b1) @ W2 + b2; scatter-add at col.
// 128 edges per block, 8 waves; 2 WMMA GEMM stages.
// LDS (64 KB): sHid 32 KB + phased sBuf 32 KB
//   phase 1: sBuf[0..2048)    = input tile, packed pairs [128 slots][16 pairs]
//            sBuf[2048..4096) = eW1 (padded to K=32), fragment-major [8][32][8]
//   phase 2: sBuf[0..8192)    = eW2, fragment-major [32][32][8]
// ---------------------------------------------------------------------------
__global__ void __launch_bounds__(256)
edge_mlp_kernel(const float* __restrict__ x, const int* __restrict__ row,
                const int* __restrict__ col, const float* __restrict__ ea,
                const float* __restrict__ eW1, const float* __restrict__ eb1,
                const float* __restrict__ eW2, const float* __restrict__ eb2,
                float* __restrict__ h, int E) {
    __shared__ __align__(16) __bf16       sHid[128 * HID];  // 32 KB
    __shared__ __align__(16) unsigned int sBuf[8192];       // 32 KB, phased

    const int tid = threadIdx.x;
    const int wave = tid >> 5, lane = tid & 31;
    const int e0 = blockIdx.x * 128;

    unsigned int* sIn = sBuf;         // [128][16] packed pairs
    unsigned int* sW1 = sBuf + 2048;  // fragment-major [8 tiles][32][8]

    // Phase-1 staging: input tile (concat features, pad to 32) + padded eW1
    for (int idx = tid; idx < 128 * 16; idx += 256) {
        int slot = idx >> 4, fp = idx & 15;
        int e = e0 + slot;
        float v0 = 0.f, v1 = 0.f;
        if (e < E) {
            int f0 = 2 * fp, f1 = 2 * fp + 1;
            int ci = col[e], ri = row[e];
            v0 = (f0 < 7) ? x[(long)ci * 7 + f0]
               : (f0 < 14) ? x[(long)ri * 7 + (f0 - 7)]
               : (f0 < 18) ? ea[(long)e * 4 + (f0 - 14)] : 0.f;
            v1 = (f1 < 7) ? x[(long)ci * 7 + f1]
               : (f1 < 14) ? x[(long)ri * 7 + (f1 - 7)]
               : (f1 < 18) ? ea[(long)e * 4 + (f1 - 14)] : 0.f;
        }
        sIn[idx] = pack_bf16x2(v0, v1);
    }
    for (int idx = tid; idx < 16 * 128; idx += 256) {
        int n = idx & 127, pk = idx >> 7;  // pk = k/2, 0..15
        int k = 2 * pk;
        float v0 = (k     < 18) ? eW1[k * HID + n]       : 0.f;
        float v1 = (k + 1 < 18) ? eW1[(k + 1) * HID + n] : 0.f;
        sW1[b_sw_index(pk, n)] = pack_bf16x2(v0, v1);
    }
    __syncthreads();

    const int r0 = wave * 16;  // this wave's 16-edge strip inside the block
    const v8f vzero = {};

    // Layer 1: [16,32] @ [32,128] + b1, relu -> sHid
    {
        v16bf a = lds_a_frag128(sIn, 16, r0, 0, lane);
#pragma unroll
        for (int t = 0; t < 8; ++t) {
            v16bf b = lds_b_frag_sw(sW1, t, lane);
            v8f c = wmma_bf16(a, b, vzero);
            int n = t * 16 + (lane & 15);
            float bv = eb1[n];
#pragma unroll
            for (int i = 0; i < 8; ++i) {
                int m = r0 + i + ((lane >= 16) ? 8 : 0);
                sHid[m * HID + n] = f2bf(fmaxf(c[i] + bv, 0.f));
            }
        }
    }
    __syncthreads();  // sHid complete; sIn/sW1 dead

    // Phase-2 staging: eW2 fragment-major
    for (int idx = tid; idx < 64 * 128; idx += 256) {
        int n = idx & 127, pk = idx >> 7;
        int k = 2 * pk;
        sBuf[b_sw_index(pk, n)] =
            pack_bf16x2(eW2[k * HID + n], eW2[(k + 1) * HID + n]);
    }
    __syncthreads();

    // Layer 2: [16,128] @ [128,128]
    v8f acc[8];
#pragma unroll
    for (int t = 0; t < 8; ++t) acc[t] = vzero;
#pragma unroll
    for (int ks = 0; ks < 4; ++ks) {
        v16bf a = lds_a_frag128((const unsigned int*)sHid, HID / 2, r0,
                                ks * 32, lane);
#pragma unroll
        for (int t = 0; t < 8; ++t) {
            v16bf b = lds_b_frag_sw(sBuf, ks * 8 + t, lane);
            acc[t] = wmma_bf16(a, b, acc[t]);
        }
    }

    // Epilogue: + eb2, scatter-add at target node (guarded for e < E)
#pragma unroll
    for (int i = 0; i < 8; ++i) {
        int m = i + ((lane >= 16) ? 8 : 0);
        int e = e0 + r0 + m;
        if (e < E) {
            long base = (long)col[e] * HID + (lane & 15);
#pragma unroll
            for (int t = 0; t < 8; ++t) {
                float v = acc[t][i] + eb2[t * 16 + (lane & 15)];
                atomicAdd(&h[base + t * 16], v);
            }
        }
    }
}

// ---------------------------------------------------------------------------
// TAGConv dense GEMM: C = act((C +) A[nrows,128] @ W[128,128] (+ bias))
// 128 rows per block, 8 waves, weights staged fragment-major in LDS.
// ---------------------------------------------------------------------------
__global__ void __launch_bounds__(256)
tag_gemm_kernel(const float* __restrict__ A, const float* __restrict__ W,
                float* __restrict__ C, const float* __restrict__ bias,
                int accumulate, int relu, int nrows) {
    __shared__ __align__(16) unsigned int sW[8192];  // 32 KB: [32 tiles][32][8]
    const int tid = threadIdx.x;
    for (int idx = tid; idx < 64 * 128; idx += 256) {
        int n = idx & 127, pk = idx >> 7;
        int k = 2 * pk;
        sW[b_sw_index(pk, n)] =
            pack_bf16x2(W[k * HID + n], W[(k + 1) * HID + n]);
    }
    __syncthreads();

    const int wave = tid >> 5, lane = tid & 31;
    const int row0 = blockIdx.x * 128 + wave * 16;
    if (row0 >= nrows) return;  // N % 16 == 0, so surviving tiles are full

    const v8f vzero = {};
    v8f acc[8];
#pragma unroll
    for (int t = 0; t < 8; ++t) acc[t] = vzero;

#pragma unroll
    for (int ks = 0; ks < 4; ++ks) {
        v16bf a = gl_a_frag_f32(A, HID, row0, ks * 32, lane);
#pragma unroll
        for (int t = 0; t < 8; ++t) {
            v16bf b = lds_b_frag_sw(sW, ks * 8 + t, lane);
            acc[t] = wmma_bf16(a, b, acc[t]);
        }
    }

#pragma unroll
    for (int t = 0; t < 8; ++t) {
        int n = t * 16 + (lane & 15);
        float bv = bias ? bias[n] : 0.f;
#pragma unroll
        for (int i = 0; i < 8; ++i) {
            int m = row0 + i + ((lane >= 16) ? 8 : 0);
            long off = (long)m * HID + n;
            float v = acc[t][i];
            if (accumulate) v += C[off];
            v += bv;
            if (relu) v = fmaxf(v, 0.f);
            C[off] = v;
        }
    }
}

// Output projection: y[n, 0:2] = h[n, :] @ out_W + out_b
__global__ void out_kernel(const float* __restrict__ h,
                           const float* __restrict__ Wo,
                           const float* __restrict__ bo,
                           float* __restrict__ y, int N) {
    int n = blockIdx.x * blockDim.x + threadIdx.x;
    if (n >= N) return;
    float s0 = bo[0], s1 = bo[1];
    const float* hr = h + (long)n * HID;
#pragma unroll 8
    for (int k = 0; k < HID; ++k) {
        float hv = hr[k];
        s0 += hv * Wo[2 * k];
        s1 += hv * Wo[2 * k + 1];
    }
    y[2 * n]     = s0;
    y[2 * n + 1] = s1;
}

// ---------------------------------------------------------------------------
// Host-side orchestration
// ---------------------------------------------------------------------------
static inline long align_up(long v, long a) { return (v + a - 1) / a * a; }

extern "C" void kernel_launch(void* const* d_in, const int* in_sizes, int n_in,
                              void* d_out, int out_size, void* d_ws, size_t ws_size,
                              hipStream_t stream) {
    const float* x    = (const float*)d_in[0];
    const int*   eidx = (const int*)d_in[1];
    const float* ea   = (const float*)d_in[2];
    const float* eW1  = (const float*)d_in[3];
    const float* eb1  = (const float*)d_in[4];
    const float* eW2  = (const float*)d_in[5];
    const float* eb2  = (const float*)d_in[6];
    const float* tagW = (const float*)d_in[7];
    const float* tagb = (const float*)d_in[8];
    const float* outW = (const float*)d_in[9];
    const float* outb = (const float*)d_in[10];

    const int N = in_sizes[0] / 7;      // 50000
    const int E = in_sizes[1] / 2;      // 600000
    const int L = 4;

    const int* row = eidx;       // edge_index[0]
    const int* col = eidx + E;   // edge_index[1]

    float* ws = (float*)d_ws;
    long off = 0;
    float* deg  = ws + off; off += align_up(N, 256);
    float* norm = ws + off; off += align_up(E, 256);
    float* hbuf = ws + off; off += (long)N * HID;
    float* obuf = ws + off; off += (long)N * HID;
    float* pa   = ws + off; off += (long)N * HID;
    float* pb   = ws + off; off += (long)N * HID;
    (void)ws_size; (void)n_in; (void)out_size;

    const long NH = (long)N * HID;
    dim3 thr(256);

    // degrees -> dinv -> edge norms
    zero_kernel<<<1024, thr, 0, stream>>>(deg, N);
    deg_kernel<<<(E + 255) / 256, thr, 0, stream>>>(col, deg, E);
    dinv_kernel<<<(N + 255) / 256, thr, 0, stream>>>(deg, N);
    norm_kernel<<<(E + 255) / 256, thr, 0, stream>>>(row, col, deg, norm, E);

    // Edge MLP + scatter-add into h
    zero_kernel<<<2048, thr, 0, stream>>>(hbuf, NH);
    edge_mlp_kernel<<<(E + 127) / 128, thr, 0, stream>>>(
        x, row, col, ea, eW1, eb1, eW2, eb2, hbuf, E);

    const int gemm_blocks = (N + 127) / 128;
    const int spmm_blocks = (int)(((long)E * 32 + 255) / 256);

    float* h = hbuf;
    float* o = obuf;
    for (int l = 0; l < L; ++l) {
        const float* Wl = tagW + (long)l * 4 * HID * HID;
        const float* bl = tagb + (long)l * HID;
        const int last_relu = (l < L - 1) ? 1 : 0;

        // out = h @ W0
        tag_gemm_kernel<<<gemm_blocks, thr, 0, stream>>>(
            h, Wl + 0 * HID * HID, o, nullptr, 0, 0, N);
        // p1 = S h ; out += p1 @ W1
        zero_kernel<<<2048, thr, 0, stream>>>(pa, NH);
        spmm_kernel<<<spmm_blocks, thr, 0, stream>>>(row, col, norm, h, pa, E);
        tag_gemm_kernel<<<gemm_blocks, thr, 0, stream>>>(
            pa, Wl + 1 * HID * HID, o, nullptr, 1, 0, N);
        // p2 = S p1 ; out += p2 @ W2
        zero_kernel<<<2048, thr, 0, stream>>>(pb, NH);
        spmm_kernel<<<spmm_blocks, thr, 0, stream>>>(row, col, norm, pa, pb, E);
        tag_gemm_kernel<<<gemm_blocks, thr, 0, stream>>>(
            pb, Wl + 2 * HID * HID, o, nullptr, 1, 0, N);
        // p3 = S p2 ; out += p3 @ W3 + b (+relu)
        zero_kernel<<<2048, thr, 0, stream>>>(pa, NH);
        spmm_kernel<<<spmm_blocks, thr, 0, stream>>>(row, col, norm, pb, pa, E);
        tag_gemm_kernel<<<gemm_blocks, thr, 0, stream>>>(
            pa, Wl + 3 * HID * HID, o, bl, 1, last_relu, N);

        // next layer input is this layer's output
        float* t = h; h = o; o = t;
    }

    out_kernel<<<(N + 255) / 256, thr, 0, stream>>>(h, outW, outb, (float*)d_out, N);
}
